// deberta_large_model_8821862826431
// MI455X (gfx1250) — matrix operand order repaired and checked by reading, not compiled, not run on hardware
//
#include <hip/hip_runtime.h>
#include <hip/hip_bf16.h>
#include <math.h>

// DeBERTa-style encoder forward on gfx1250.
// All GEMMs run on v_wmma_f32_16x16x32_bf16 (fp32 -> bf16 inputs, f32 accum).
// GEMM is software-pipelined: next K-slab global loads are issued before the
// current slab's WMMAs, conversion+LDS stores happen after (double-buffered).
// Workspace usage: 21M floats = 84 MB.

namespace {

constexpr int Bc = 2, Sc = 512, Lc = 12, Hc = 1024, NHc = 16, DHc = 64, FFc = 4096;
constexpr float INV_SCALE = 0.07216878364870323f;   // 1/sqrt(3*DH)
constexpr float LN_EPS = 1e-7f;

typedef __bf16 bf16;
typedef bf16 v16bf __attribute__((ext_vector_type(16)));
typedef bf16 v8bf  __attribute__((ext_vector_type(8)));
typedef bf16 v4bf  __attribute__((ext_vector_type(4)));
typedef float v8f  __attribute__((ext_vector_type(8)));
typedef float v4f  __attribute__((ext_vector_type(4)));

__device__ __forceinline__ v16bf cat8(v8bf lo, v8bf hi) {
  return __builtin_shufflevector(lo, hi, 0,1,2,3,4,5,6,7,8,9,10,11,12,13,14,15);
}

// Build a 16-element bf16 fragment from two runs of 8 consecutive fp32 values.
__device__ __forceinline__ v16bf frag_from_f32(const float* p0, const float* p1) {
  v4f a0 = *(const v4f*)p0;
  v4f a1 = *(const v4f*)(p0 + 4);
  v4f b0 = *(const v4f*)p1;
  v4f b1 = *(const v4f*)(p1 + 4);
  v16bf r;
  r[0]=(bf16)a0.x;  r[1]=(bf16)a0.y;  r[2]=(bf16)a0.z;  r[3]=(bf16)a0.w;
  r[4]=(bf16)a1.x;  r[5]=(bf16)a1.y;  r[6]=(bf16)a1.z;  r[7]=(bf16)a1.w;
  r[8]=(bf16)b0.x;  r[9]=(bf16)b0.y;  r[10]=(bf16)b0.z; r[11]=(bf16)b0.w;
  r[12]=(bf16)b1.x; r[13]=(bf16)b1.y; r[14]=(bf16)b1.z; r[15]=(bf16)b1.w;
  return r;
}

__device__ __forceinline__ v4bf cvt4(v4f v) {
  v4bf t; t[0]=(bf16)v.x; t[1]=(bf16)v.y; t[2]=(bf16)v.z; t[3]=(bf16)v.w;
  return t;
}

__device__ __forceinline__ float gelu_tanh(float x) {
  return 0.5f * x * (1.f + tanhf(0.7978845608028654f * (x + 0.044715f * x * x * x)));
}

// ---------------------------------------------------------------------------
// Generic batched GEMM: C[z] = A[z](MxK) * W[z](NxK)^T (+bias) (gelu?) *oscale
// fp32 operands, bf16 WMMA with f32 accumulate.
// Batch z decomposes as (zb, zh) with independent strides (per-head views into
// the packed QKV buffer). Requires M%128==0, N%64==0, K%32==0, 16B alignment.
// ---------------------------------------------------------------------------
constexpr int BM = 128, BN = 64, BK = 32;

__global__ __launch_bounds__(256)
void gemm_bf16(const float* __restrict__ A, long long sAb, long long sAh, int lda,
               const float* __restrict__ W, long long sWb, long long sWh, int ldw,
               float* __restrict__ C, long long sCb, long long sCh, int ldc,
               int K, const float* __restrict__ bias, float oscale, int act, int nh)
{
  __shared__ __align__(16) bf16 As[2][BM * BK];
  __shared__ __align__(16) bf16 Ws[2][BN * BK];

  const int z  = blockIdx.z;
  const int zb = z / nh, zh = z % nh;
  A += (size_t)zb * sAb + (size_t)zh * sAh;
  W += (size_t)zb * sWb + (size_t)zh * sWh;
  C += (size_t)zb * sCb + (size_t)zh * sCh;

  const int m0 = blockIdx.y * BM;
  const int n0 = blockIdx.x * BN;
  const int tid  = threadIdx.x;
  const int lane = tid & 31;
  const int wave = tid >> 5;
  const int wm = (wave >> 1) * 32;   // 0,32,64,96
  const int wn = (wave & 1) * 32;    // 0,32

  // per-thread staging coordinates (A: 4 float4 / thread, W: 2 float4 / thread)
  size_t aoff[4], woff[2];
  int    aslot[4], wslot[2];
  #pragma unroll
  for (int j = 0; j < 4; ++j) {
    int e = tid + j * 256;           // float4 index, 8 per 32-float row
    int m = e >> 3, k = (e & 7) * 4;
    aoff[j]  = (size_t)(m0 + m) * lda + k;
    aslot[j] = m * BK + k;
  }
  #pragma unroll
  for (int j = 0; j < 2; ++j) {
    int e = tid + j * 256;
    int n = e >> 3, k = (e & 7) * 4;
    woff[j]  = (size_t)(n0 + n) * ldw + k;
    wslot[j] = n * BK + k;
  }

  const int off8  = (lane >> 4) * 8;    // A: K sub-offset per half-wave
  const int off16 = (lane >> 4) * 16;   // unused; B is read with off8 like A
  const int arow  = wm + (lane & 15);
  const int brow  = wn + (lane & 15);

  v8f acc[2][2] = {};
  const int nk = K / BK;

  // ---- preload tile 0 into registers, then LDS buffer 0 ----
  v4f areg[4], wreg[2];
  #pragma unroll
  for (int j = 0; j < 4; ++j) areg[j] = *(const v4f*)(A + aoff[j]);
  #pragma unroll
  for (int j = 0; j < 2; ++j) wreg[j] = *(const v4f*)(W + woff[j]);
  #pragma unroll
  for (int j = 0; j < 4; ++j) *(v4bf*)(&As[0][aslot[j]]) = cvt4(areg[j]);
  #pragma unroll
  for (int j = 0; j < 2; ++j) *(v4bf*)(&Ws[0][wslot[j]]) = cvt4(wreg[j]);
  __syncthreads();

  for (int ks = 0; ks < nk; ++ks) {
    const int cur = ks & 1;
    const bool more = (ks + 1 < nk);

    // issue ALL next-slab global loads up front (latency hidden under WMMA)
    if (more) {
      const size_t kadv = (size_t)(ks + 1) * BK;
      #pragma unroll
      for (int j = 0; j < 4; ++j) areg[j] = *(const v4f*)(A + aoff[j] + kadv);
      #pragma unroll
      for (int j = 0; j < 2; ++j) wreg[j] = *(const v4f*)(W + woff[j] + kadv);
      if (ks + 2 < nk) {   // L2 prefetch two slabs ahead (global_prefetch_b8)
        __builtin_prefetch(A + aoff[0] + kadv + BK, 0, 3);
        __builtin_prefetch(W + woff[0] + kadv + BK, 0, 3);
      }
    }

    // ---- compute current slab from LDS ----
    v16bf afr[2], wfr[2];
    #pragma unroll
    for (int mf = 0; mf < 2; ++mf) {
      const bf16* p = &As[cur][(arow + mf * 16) * BK];
      v8bf lo = *(const v8bf*)(p + off8);        // K = off8 .. off8+7
      v8bf hi = *(const v8bf*)(p + 16 + off8);   // K = 16+off8 .. +7
      afr[mf] = cat8(lo, hi);
    }
    #pragma unroll
    for (int nf = 0; nf < 2; ++nf) {
      const bf16* p = &Ws[cur][(brow + nf * 16) * BK];
      v8bf lo = *(const v8bf*)(p + off8);
      v8bf hi = *(const v8bf*)(p + 16 + off8);
      wfr[nf] = cat8(lo, hi);
    }
    #pragma unroll
    for (int mf = 0; mf < 2; ++mf)
      #pragma unroll
      for (int nf = 0; nf < 2; ++nf)
        acc[mf][nf] = __builtin_amdgcn_wmma_f32_16x16x32_bf16(
            false, afr[mf], false, wfr[nf], (short)0, acc[mf][nf], false, false);

    // ---- convert + store next slab into the other LDS buffer ----
    if (more) {
      const int nxt = cur ^ 1;
      #pragma unroll
      for (int j = 0; j < 4; ++j) *(v4bf*)(&As[nxt][aslot[j]]) = cvt4(areg[j]);
      #pragma unroll
      for (int j = 0; j < 2; ++j) *(v4bf*)(&Ws[nxt][wslot[j]]) = cvt4(wreg[j]);
      __syncthreads();
    }
  }

  // ---- epilogue: C/D layout -> (m = vgpr + 8*(lane>=16), n = lane&15) ----
  const int mbase = (lane >> 4) * 8;
  const int ncol  = lane & 15;
  #pragma unroll
  for (int mf = 0; mf < 2; ++mf)
    #pragma unroll
    for (int nf = 0; nf < 2; ++nf) {
      int col = n0 + wn + nf * 16 + ncol;
      float bv = bias ? bias[col] : 0.f;
      #pragma unroll
      for (int i = 0; i < 8; ++i) {
        int row = m0 + wm + mf * 16 + mbase + i;
        float v = acc[mf][nf][i] + bv;
        if (act) v = gelu_tanh(v);
        v *= oscale;
        C[(size_t)row * ldc + col] = v;
      }
    }
}

// ---------------------------------------------------------------------------
// Relative-position terms via WMMA + Toeplitz scatter-add.
// mode 0 (c2p): rows = q (scaled Q part of qkv), P = pk,
//               scores[q, q-r+512] += q[q,:].pk[r,:]
// mode 1 (p2c): rows = k (K part of qkv), P = pq (scaled),
//               scores[k+r-512, k] += k[k,:].pq[r,:]
// 4 waves per block; each wave computes one 16x16 (row x r) tile, K = DH = 64.
// Out-of-band tiles early-exit (wave-uniform, EXEC stays all-ones for WMMA).
// ---------------------------------------------------------------------------
__global__ __launch_bounds__(128)
void relpos_scatter(const float* __restrict__ qkv, const float* __restrict__ P,
                    float* __restrict__ scores, int mode)
{
  const int wave = threadIdx.x >> 5;
  const int lane = threadIdx.x & 31;
  const int r0 = (blockIdx.x * 4 + wave) * 16;
  const int m0 = blockIdx.y * 16;
  const int bh = blockIdx.z;
  const int b = bh >> 4, h = bh & 15;

  if (mode == 0) {                     // k = m - r + 512
    if (m0 + 15 - r0 + 512 < 0) return;
    if (m0 - (r0 + 15) + 512 > Sc - 1) return;
  } else {                             // q = m + r - 512
    if (m0 + 15 + r0 + 15 - 512 < 0) return;
    if (m0 + r0 - 512 > Sc - 1) return;
  }

  const float* Arow = qkv + (size_t)(b * Sc + m0 + (lane & 15)) * (3 * Hc)
                          + h * (3 * DHc) + (mode ? DHc : 0);
  const float* Prow = P + (size_t)(r0 + (lane & 15)) * Hc + h * DHc;
  const int off8  = (lane >> 4) * 8;
  const int off16 = (lane >> 4) * 16;

  v8f acc = {};
  #pragma unroll
  for (int ks = 0; ks < 2; ++ks) {
    const int kb = ks * 32;
    v16bf a = frag_from_f32(Arow + kb + off8, Arow + kb + 16 + off8);
    v16bf w = frag_from_f32(Prow + kb + off8, Prow + kb + 16 + off8);
    acc = __builtin_amdgcn_wmma_f32_16x16x32_bf16(false, a, false, w,
                                                  (short)0, acc, false, false);
  }

  float* sp = scores + (size_t)bh * Sc * Sc;
  const int mrow = m0 + (lane >> 4) * 8;
  const int rr   = r0 + (lane & 15);
  #pragma unroll
  for (int i = 0; i < 8; ++i) {
    int m = mrow + i;
    if (mode == 0) {
      int k = m - rr + 512;
      if (k >= 0 && k < Sc) sp[(size_t)m * Sc + k] += acc[i];
    } else {
      int q = m + rr - 512;
      if (q >= 0 && q < Sc) sp[(size_t)q * Sc + m] += acc[i];
    }
  }
}

// ---------------------------------------------------------------------------
// Fold q_bias (+1/scale) into Q in place; build bias-added V^T [b,h,d,s].
// ---------------------------------------------------------------------------
__global__ __launch_bounds__(256)
void qkv_bias_vt(float* __restrict__ qkv, float* __restrict__ vT,
                 const float* __restrict__ qb, const float* __restrict__ vb)
{
  int i  = blockIdx.x * 256 + threadIdx.x;     // < B*S*H
  int bs = i >> 10;
  int hd = i & 1023;
  int nh = hd >> 6, d = hd & 63;
  size_t base = (size_t)bs * (3 * Hc) + nh * (3 * DHc);
  qkv[base + d] = (qkv[base + d] + qb[hd]) * INV_SCALE;
  float v = qkv[base + 2 * DHc + d] + vb[hd];
  int b = bs >> 9, s = bs & 511;
  vT[((size_t)((b * NHc + nh) * DHc + d)) * Sc + s] = v;
}

// ---------------------------------------------------------------------------
// Row softmax over K with padding mask; probs *= m[q]*m[k]. In-place.
// ---------------------------------------------------------------------------
__global__ __launch_bounds__(256)
void softmax_mask(float* __restrict__ scores, const int* __restrict__ mask)
{
  __shared__ float red[256];
  const int row = blockIdx.x;                 // bh*512 + q
  const int q  = row & 511;
  const int bh = row >> 9;
  const int b  = bh >> 4;
  float* sp = scores + (size_t)row * Sc;
  const int tid = threadIdx.x;
  const int mq  = mask[b * Sc + q];
  const int mk0 = mask[b * Sc + tid];
  const int mk1 = mask[b * Sc + tid + 256];
  float x0 = (mq > 0 && mk0 > 0) ? sp[tid]       : -1e9f;
  float x1 = (mq > 0 && mk1 > 0) ? sp[tid + 256] : -1e9f;

  red[tid] = fmaxf(x0, x1); __syncthreads();
  for (int s = 128; s > 0; s >>= 1) { if (tid < s) red[tid] = fmaxf(red[tid], red[tid + s]); __syncthreads(); }
  const float mx = red[0]; __syncthreads();

  float e0 = __expf(x0 - mx), e1 = __expf(x1 - mx);
  red[tid] = e0 + e1; __syncthreads();
  for (int s = 128; s > 0; s >>= 1) { if (tid < s) red[tid] += red[tid + s]; __syncthreads(); }
  const float inv = 1.f / red[0];

  sp[tid]       = e0 * inv * (float)(mq * mk0);
  sp[tid + 256] = e1 * inv * (float)(mq * mk1);
}

// ---------------------------------------------------------------------------
// Embedding gather + LayerNorm * mask.
// ---------------------------------------------------------------------------
__global__ __launch_bounds__(256)
void embed_ln(const int* __restrict__ ids, const int* __restrict__ mask,
              const float* __restrict__ we, const float* __restrict__ ls,
              const float* __restrict__ lb, float* __restrict__ out)
{
  __shared__ float red[256];
  const int row = blockIdx.x;
  const int tid = threadIdx.x;
  const float* src = we + (size_t)ids[row] * Hc;
  float x[4];
  #pragma unroll
  for (int j = 0; j < 4; ++j) x[j] = src[tid + j * 256];

  red[tid] = x[0] + x[1] + x[2] + x[3]; __syncthreads();
  for (int s = 128; s > 0; s >>= 1) { if (tid < s) red[tid] += red[tid + s]; __syncthreads(); }
  const float mu = red[0] * (1.f / Hc); __syncthreads();

  float vs = 0.f;
  #pragma unroll
  for (int j = 0; j < 4; ++j) { float d = x[j] - mu; vs += d * d; }
  red[tid] = vs; __syncthreads();
  for (int s = 128; s > 0; s >>= 1) { if (tid < s) red[tid] += red[tid + s]; __syncthreads(); }
  const float rs = rsqrtf(red[0] * (1.f / Hc) + LN_EPS);
  const float mfac = (float)mask[row];
  #pragma unroll
  for (int j = 0; j < 4; ++j) {
    int c = tid + j * 256;
    out[(size_t)row * Hc + c] = ((x[j] - mu) * rs * ls[c] + lb[c]) * mfac;
  }
}

// ---------------------------------------------------------------------------
// hc = LN(hc + add)  (row per block)
// ---------------------------------------------------------------------------
__global__ __launch_bounds__(256)
void ln_residual(float* __restrict__ hc, const float* __restrict__ add,
                 const float* __restrict__ ls, const float* __restrict__ lb)
{
  __shared__ float red[256];
  const int row = blockIdx.x;
  const int tid = threadIdx.x;
  float x[4];
  #pragma unroll
  for (int j = 0; j < 4; ++j) {
    size_t idx = (size_t)row * Hc + tid + j * 256;
    x[j] = hc[idx] + add[idx];
  }
  red[tid] = x[0] + x[1] + x[2] + x[3]; __syncthreads();
  for (int s = 128; s > 0; s >>= 1) { if (tid < s) red[tid] += red[tid + s]; __syncthreads(); }
  const float mu = red[0] * (1.f / Hc); __syncthreads();
  float vs = 0.f;
  #pragma unroll
  for (int j = 0; j < 4; ++j) { float d = x[j] - mu; vs += d * d; }
  red[tid] = vs; __syncthreads();
  for (int s = 128; s > 0; s >>= 1) { if (tid < s) red[tid] += red[tid + s]; __syncthreads(); }
  const float rs = rsqrtf(red[0] * (1.f / Hc) + LN_EPS);
  #pragma unroll
  for (int j = 0; j < 4; ++j) {
    int c = tid + j * 256;
    hc[(size_t)row * Hc + c] = (x[j] - mu) * rs * ls[c] + lb[c];
  }
}

// ---------------------------------------------------------------------------
// Head: out[b] = LN(hc[b,0,:], eps=1e-5) . out_w + out_b
// ---------------------------------------------------------------------------
__global__ __launch_bounds__(256)
void head_kernel(const float* __restrict__ hc, const float* __restrict__ ls,
                 const float* __restrict__ lb, const float* __restrict__ ow,
                 const float* __restrict__ ob, float* __restrict__ out)
{
  __shared__ float red[256];
  const int b = blockIdx.x;
  const int tid = threadIdx.x;
  const float* src = hc + (size_t)b * Sc * Hc;   // token 0
  float x[4];
  #pragma unroll
  for (int j = 0; j < 4; ++j) x[j] = src[tid + j * 256];
  red[tid] = x[0] + x[1] + x[2] + x[3]; __syncthreads();
  for (int s = 128; s > 0; s >>= 1) { if (tid < s) red[tid] += red[tid + s]; __syncthreads(); }
  const float mu = red[0] * (1.f / Hc); __syncthreads();
  float vs = 0.f;
  #pragma unroll
  for (int j = 0; j < 4; ++j) { float d = x[j] - mu; vs += d * d; }
  red[tid] = vs; __syncthreads();
  for (int s = 128; s > 0; s >>= 1) { if (tid < s) red[tid] += red[tid + s]; __syncthreads(); }
  const float rs = rsqrtf(red[0] * (1.f / Hc) + 1e-5f); __syncthreads();
  float dot = 0.f;
  #pragma unroll
  for (int j = 0; j < 4; ++j) {
    int c = tid + j * 256;
    dot += ((x[j] - mu) * rs * ls[c] + lb[c]) * ow[c];
  }
  red[tid] = dot; __syncthreads();
  for (int s = 128; s > 0; s >>= 1) { if (tid < s) red[tid] += red[tid + s]; __syncthreads(); }
  if (tid == 0) out[b] = red[0] + ob[0];
}

} // namespace

// ---------------------------------------------------------------------------
extern "C" void kernel_launch(void* const* d_in, const int* in_sizes, int n_in,
                              void* d_out, int out_size, void* d_ws, size_t ws_size,
                              hipStream_t stream)
{
  (void)in_sizes; (void)n_in; (void)out_size; (void)ws_size;

  const int*   ids       = (const int*)  d_in[0];
  const int*   mask      = (const int*)  d_in[1];
  const float* word_emb  = (const float*)d_in[3];
  const float* emb_ln_s  = (const float*)d_in[4];
  const float* emb_ln_b  = (const float*)d_in[5];
  const float* rel_emb   = (const float*)d_in[6];
  const float* in_w      = (const float*)d_in[7];
  const float* q_bias    = (const float*)d_in[8];
  const float* v_bias    = (const float*)d_in[9];
  const float* pos_k_w   = (const float*)d_in[10];
  const float* pos_q_w   = (const float*)d_in[11];
  const float* pos_q_b   = (const float*)d_in[12];
  const float* attn_o_w  = (const float*)d_in[13];
  const float* attn_o_b  = (const float*)d_in[14];
  const float* ln1_s     = (const float*)d_in[15];
  const float* ln1_b     = (const float*)d_in[16];
  const float* ffn_w1    = (const float*)d_in[17];
  const float* ffn_b1    = (const float*)d_in[18];
  const float* ffn_w2    = (const float*)d_in[19];
  const float* ffn_b2    = (const float*)d_in[20];
  const float* ln2_s     = (const float*)d_in[21];
  const float* ln2_b     = (const float*)d_in[22];
  const float* head_ln_s = (const float*)d_in[23];
  const float* head_ln_b = (const float*)d_in[24];
  const float* out_w     = (const float*)d_in[25];
  const float* out_b     = (const float*)d_in[26];
  float* out = (float*)d_out;

  // workspace layout (floats)
  const size_t M1 = 1u << 20;
  float* ws     = (float*)d_ws;
  float* hc     = ws;                 // 1M  [B*S, H]
  float* tmp    = ws + 1 * M1;        // 1M  [B*S, H]
  float* qkv    = ws + 2 * M1;        // 3M  [B*S, 3H]
  float* vT     = ws + 5 * M1;        // 1M  [B, NH, DH, S]
  float* pk     = ws + 6 * M1;        // 1M  [2*SPAN, H]
  float* pq     = ws + 7 * M1;        // 1M  [2*SPAN, H]
  float* scores = ws + 8 * M1;        // 8M  [B, NH, S, S]
  float* ctx    = ws + 16 * M1;       // 1M  [B*S, H]
  float* ff     = ws + 17 * M1;       // 4M  [B*S, FF]

  const int M = Bc * Sc;              // 1024 tokens

  // Embedding + LN + mask
  embed_ln<<<M, 256, 0, stream>>>(ids, mask, word_emb, emb_ln_s, emb_ln_b, hc);

  for (int l = 0; l < Lc; ++l) {
    const float* iw  = in_w     + (size_t)l * 3 * Hc * Hc;
    const float* pkw = pos_k_w  + (size_t)l * Hc * Hc;
    const float* pqw = pos_q_w  + (size_t)l * Hc * Hc;
    const float* aow = attn_o_w + (size_t)l * Hc * Hc;
    const float* w1  = ffn_w1   + (size_t)l * FFc * Hc;
    const float* w2  = ffn_w2   + (size_t)l * Hc * FFc;

    // QKV projection: [1024,1024] x [3072,1024]^T
    gemm_bf16<<<dim3(3 * Hc / BN, M / BM, 1), 256, 0, stream>>>(
        hc, 0, 0, Hc, iw, 0, 0, Hc, qkv, 0, 0, 3 * Hc,
        Hc, nullptr, 1.f, 0, 1);
    // fold q_bias (+1/scale) into Q, build V^T with v_bias
    qkv_bias_vt<<<(M * Hc) / 256, 256, 0, stream>>>(qkv, vT,
        q_bias + (size_t)l * Hc, v_bias + (size_t)l * Hc);

    // pos-key / pos-query projections: [1024,1024] x [1024,1024]^T
    gemm_bf16<<<dim3(Hc / BN, 2 * Sc / BM, 1), 256, 0, stream>>>(
        rel_emb, 0, 0, Hc, pkw, 0, 0, Hc, pk, 0, 0, Hc,
        Hc, nullptr, 1.f, 0, 1);
    gemm_bf16<<<dim3(Hc / BN, 2 * Sc / BM, 1), 256, 0, stream>>>(
        rel_emb, 0, 0, Hc, pqw, 0, 0, Hc, pq, 0, 0, Hc,
        Hc, pos_q_b + (size_t)l * Hc, INV_SCALE, 0, 1);

    // content-content scores: per (b,h)  q[512,64] x k[512,64]^T
    gemm_bf16<<<dim3(Sc / BN, Sc / BM, Bc * NHc), 256, 0, stream>>>(
        qkv,       (long long)Sc * 3 * Hc, 3 * DHc, 3 * Hc,
        qkv + DHc, (long long)Sc * 3 * Hc, 3 * DHc, 3 * Hc,
        scores, (long long)NHc * Sc * Sc, (long long)Sc * Sc, Sc,
        DHc, nullptr, 1.f, 0, NHc);

    // c2p / p2c relative-position scatter-adds (4 r-tiles per block)
    relpos_scatter<<<dim3(2 * Sc / 64, Sc / 16, Bc * NHc), 128, 0, stream>>>(
        qkv, pk, scores, 0);
    relpos_scatter<<<dim3(2 * Sc / 64, Sc / 16, Bc * NHc), 128, 0, stream>>>(
        qkv, pq, scores, 1);

    // masked softmax (in place)
    softmax_mask<<<Bc * NHc * Sc, 256, 0, stream>>>(scores, mask);

    // context: per (b,h)  probs[512,512] x vT[64,512]^T -> ctx[b, q, h*64+d]
    gemm_bf16<<<dim3(DHc / BN, Sc / BM, Bc * NHc), 256, 0, stream>>>(
        scores, (long long)NHc * Sc * Sc, (long long)Sc * Sc, Sc,
        vT,     (long long)NHc * DHc * Sc, (long long)DHc * Sc, Sc,
        ctx,    (long long)Sc * Hc, DHc, Hc,
        Sc, nullptr, 1.f, 0, NHc);

    // attention output projection
    gemm_bf16<<<dim3(Hc / BN, M / BM, 1), 256, 0, stream>>>(
        ctx, 0, 0, Hc, aow, 0, 0, Hc, tmp, 0, 0, Hc,
        Hc, attn_o_b + (size_t)l * Hc, 1.f, 0, 1);
    ln_residual<<<M, 256, 0, stream>>>(hc, tmp,
        ln1_s + (size_t)l * Hc, ln1_b + (size_t)l * Hc);

    // FFN
    gemm_bf16<<<dim3(FFc / BN, M / BM, 1), 256, 0, stream>>>(
        hc, 0, 0, Hc, w1, 0, 0, Hc, ff, 0, 0, FFc,
        Hc, ffn_b1 + (size_t)l * FFc, 1.f, /*gelu*/1, 1);
    gemm_bf16<<<dim3(Hc / BN, M / BM, 1), 256, 0, stream>>>(
        ff, 0, 0, FFc, w2, 0, 0, FFc, tmp, 0, 0, Hc,
        FFc, ffn_b2 + (size_t)l * Hc, 1.f, 0, 1);
    ln_residual<<<M, 256, 0, stream>>>(hc, tmp,
        ln2_s + (size_t)l * Hc, ln2_b + (size_t)l * Hc);
  }

  head_kernel<<<Bc, 256, 0, stream>>>(hc, head_ln_s, head_ln_b, out_w, out_b, out);
}
